// Update_Edge_80711025426902
// MI455X (gfx1250) — matrix operand order repaired and checked by reading, not compiled, not run on hardware
//
#include <hip/hip_runtime.h>

// ---------------------------------------------------------------------------
// AF2-style outer-product-mean edge update for MI455X (gfx1250, wave32, WMMA)
// N=1, D=128, L=256, C=256, OC=32, PC=256
//
// k1: LayerNorm + dual projection -> left_t/right_t bf16 [L][OC][D] in ws
// k2: fused (outer-product over d) x (output projection) using
//     v_wmma_f32_16x16x32_bf16; O-tile staged in LDS between the two GEMMs;
//     Wout tiles streamed into LDS by the Tensor Data Mover (double-buffered,
//     TENSORcnt-tracked), overlapping DMA with WMMA compute.
// ---------------------------------------------------------------------------

#define L_DIM 256
#define D_DIM 128
#define C_DIM 256
#define OC    32
#define PC    256
#define KDIM  (OC * OC)   // 1024

typedef __attribute__((ext_vector_type(16))) __bf16 v16bf;
typedef __attribute__((ext_vector_type(8)))  float  v8f;
typedef unsigned int u32x4 __attribute__((ext_vector_type(4)));
typedef int          i32x4 __attribute__((ext_vector_type(4)));
typedef int          i32x8 __attribute__((ext_vector_type(8)));

struct U128 { unsigned int x, y, z, w; };
struct U256 { U128 a, b; };

#if defined(__has_builtin)
#if __has_builtin(__builtin_amdgcn_tensor_load_to_lds) && \
    __has_builtin(__builtin_amdgcn_s_wait_tensorcnt)
#define HAVE_TDM 1
#else
#define HAVE_TDM 0
#endif
#else
#define HAVE_TDM 0
#endif

__device__ inline unsigned short f32_to_bf16(float f) {
    unsigned int u = __builtin_bit_cast(unsigned int, f);
    u += 0x7FFFu + ((u >> 16) & 1u);   // round-to-nearest-even
    return (unsigned short)(u >> 16);
}

// Load a 16-element bf16 fragment from two 16-byte chunks (8 + 8 elements).
__device__ inline v16bf load_frag(const unsigned short* p0, const unsigned short* p1) {
    U256 t;
    t.a = *(const U128*)p0;
    t.b = *(const U128*)p1;
    return __builtin_bit_cast(v16bf, t);
}

// ---------------------------------------------------------------------------
// TDM: DMA one Wout tile (rows p=0..255, cols q=tile*32..tile*32+31, bf16)
// from global into LDS (row-major [p][32q], 16 KB). Issued by wave 0 only;
// completion tracked with TENSORcnt. Fallback: cooperative vector copy.
// ---------------------------------------------------------------------------
__device__ inline void load_wout_tile(unsigned short* lds_dst,
                                      const unsigned short* wout_bf,
                                      int tile, int wave) {
#if HAVE_TDM
    if (wave == 0) {
        const unsigned short* g = wout_bf + tile * 32;
        unsigned long long ga = (unsigned long long)(size_t)g;
        unsigned int lds_addr = (unsigned int)(size_t)(void*)lds_dst;
        u32x4 g0;
        g0.x = 1u;                          // count=1, user descriptor
        g0.y = lds_addr;                    // lds_addr (bytes)
        g0.z = (unsigned int)ga;            // global_addr[31:0]
        g0.w = (unsigned int)((ga >> 32) & 0x01FFFFFFu) | 0x80000000u; // [56:32], type=2
        i32x8 g1 = { 0x00010000,            // data_size = 2 bytes
                     0x04000000,            // tensor_dim0 = 1024
                     0x01000000,            // tensor_dim1 = 256
                     0x00200000,            // tile_dim0 = 32
                     0x00000100,            // tile_dim1 = 256 (tile_dim2 = 0)
                     0x00000400,            // tensor_dim0_stride = 1024
                     0, 0 };
        i32x4 gz = {0, 0, 0, 0};            // groups 2/3: dims unused (2-D tile)
#if __clang_major__ >= 23
        i32x8 gz8 = {0, 0, 0, 0, 0, 0, 0, 0};
        __builtin_amdgcn_tensor_load_to_lds(g0, g1, gz, gz, gz8, 0);
#else
        __builtin_amdgcn_tensor_load_to_lds(g0, g1, gz, gz, 0);
#endif
    }
#else
    const unsigned short* src = wout_bf + tile * 32;
    for (int i = threadIdx.x; i < 1024; i += 256) {   // 256x32 bf16 = 16 KB
        int row = i >> 2, part = i & 3;
        *(U128*)(lds_dst + row * 32 + part * 8) =
            *(const U128*)(src + (size_t)row * KDIM + part * 8);
    }
#endif
}

__device__ inline void wait_wout_tile(int wave) {
#if HAVE_TDM
    if (wave == 0) __builtin_amdgcn_s_wait_tensorcnt(0);
#endif
}

// ---------------------------------------------------------------------------
// Kernel 0: convert Wout f32 [PC][KDIM] -> bf16 (row-major, same layout)
// ---------------------------------------------------------------------------
__global__ __launch_bounds__(256) void k_cvt_wout(const float* __restrict__ w,
                                                  unsigned short* __restrict__ wbf,
                                                  int n) {
    int i = blockIdx.x * 256 + threadIdx.x;
    if (i < n) wbf[i] = f32_to_bf16(w[i]);
}

// ---------------------------------------------------------------------------
// Kernel 1: LayerNorm over C + left/right projections.
// One wave32 per (d,l) row. Outputs transposed: left_t/right_t [L][OC][D] bf16.
// right is pre-scaled by 1/D.
// ---------------------------------------------------------------------------
__global__ __launch_bounds__(256) void k_ln_proj(
    const float* __restrict__ msa,   // [D][L][C]
    const float* __restrict__ ln_w,  // [C]
    const float* __restrict__ ln_b,  // [C]
    const float* __restrict__ Wl,    // [OC][C]
    const float* __restrict__ bl,    // [OC]
    const float* __restrict__ Wr,    // [OC][C]
    const float* __restrict__ br,    // [OC]
    unsigned short* __restrict__ left_t,   // [L][OC][D]
    unsigned short* __restrict__ right_t)  // [L][OC][D]
{
    __shared__ float xs[8][C_DIM];
    const int lane = threadIdx.x & 31;
    const int wave = threadIdx.x >> 5;
    const int row  = blockIdx.x * 8 + wave;    // row = d*L + l
    const int d    = row >> 8;
    const int l    = row & (L_DIM - 1);

    const float* x = msa + (size_t)row * C_DIM;
    float v[8], s = 0.f, s2 = 0.f;
#pragma unroll
    for (int i = 0; i < 8; ++i) {
        float t = x[lane + 32 * i];
        v[i] = t; s += t; s2 += t * t;
    }
#pragma unroll
    for (int off = 16; off >= 1; off >>= 1) {     // wave32 reduction
        s  += __shfl_xor(s,  off, 32);
        s2 += __shfl_xor(s2, off, 32);
    }
    const float mu   = s * (1.f / C_DIM);
    const float var  = s2 * (1.f / C_DIM) - mu * mu;
    const float rstd = rsqrtf(var + 1e-5f);
#pragma unroll
    for (int i = 0; i < 8; ++i) {
        int c = lane + 32 * i;
        xs[wave][c] = (v[i] - mu) * rstd * ln_w[c] + ln_b[c];
    }
    __syncthreads();

    const int o = lane;                           // one output channel per lane
    const float* wl = Wl + o * C_DIM;
    const float* wr = Wr + o * C_DIM;
    float aL = bl[o], aR = br[o];
#pragma unroll 4
    for (int c = 0; c < C_DIM; ++c) {
        float xv = xs[wave][c];
        aL = fmaf(xv, wl[c], aL);
        aR = fmaf(xv, wr[c], aR);
    }
    aR *= (1.0f / D_DIM);
    left_t [((size_t)l * OC + o) * D_DIM + d] = f32_to_bf16(aL);
    right_t[((size_t)l * OC + o) * D_DIM + d] = f32_to_bf16(aR);
}

// ---------------------------------------------------------------------------
// Kernel 2: fused outer-product (sum over d) + output projection.
// Block = 256 threads = 8 waves; block owns a 4x4 (l,m) tile -> out[16 lm][256 p].
//
// Stage A: O[(li,c),(mi,q)] = sum_d left[d,l,c] * right[d,m,q]
//   M-tile = (li, 16 c's), N-tile = (mi, 16 q's), K = d (4 chunks of 32).
//   Results stored bf16 into LDS O[li][c][mi][q] (32 KB).
//   Meanwhile the TDM streams Wout tile 0 into LDS (overlaps all of stage A).
// Stage B: out[(li,mi), p] = sum_{c,q} O * WoutT[(c,q), p]
//   Single M-tile of 16 lm rows; K = c*32+q in 32 chunks; wave w owns p-slice
//   [32w, 32w+32) -> two 16x16 accumulators. Wout chunk c+1 is TDM-prefetched
//   into the other LDS buffer while chunk c is consumed.
// ---------------------------------------------------------------------------
__global__ __launch_bounds__(256) void k_outer_proj(
    const unsigned short* __restrict__ left_t,   // [L][OC][D]
    const unsigned short* __restrict__ right_t,  // [L][OC][D]
    const unsigned short* __restrict__ wout_bf,  // [PC][KDIM]
    const float* __restrict__ bout,              // [PC]
    float* __restrict__ out)                     // [L][L][PC]
{
    __shared__ unsigned short O[4 * 32 * 4 * 32];     // [li][c][mi][q], 32 KB
    __shared__ unsigned short WTILE[2][PC * 32];      // Wout tiles, 2 x 16 KB

    const int lane = threadIdx.x & 31;
    const int wave = threadIdx.x >> 5;
    const int hi   = lane >> 4;          // 0: lanes 0-15, 1: lanes 16-31
    const int lrow = lane & 15;
    const int tile = blockIdx.x;
    const int l0 = (tile >> 6) * 4;
    const int m0 = (tile & 63) * 4;

    // Kick off DMA of Wout tile 0 (c = 0); overlaps all of stage A.
    load_wout_tile(WTILE[0], wout_bf, 0, wave);

    // ----------------- Stage A: outer-product GEMM over d -----------------
    {
        const int mt = wave;                  // wave -> one of 8 M-tiles
        const int li = mt >> 1;
        const int c0 = (mt & 1) * 16;

        // A fragments (16x32 bf16): row = c0+lrow, K = d chunk.
        // lanes 0-15 hold K 0..15; lanes 16-31 hold K 16..31.
        const unsigned short* abase =
            left_t + ((size_t)(l0 + li) * OC + (c0 + lrow)) * D_DIM + hi * 16;
        v16bf afrag[4];
#pragma unroll
        for (int kc = 0; kc < 4; ++kc) {
            const unsigned short* p = abase + kc * 32;
            afrag[kc] = load_frag(p, p + 8);
        }

#pragma unroll
        for (int nt = 0; nt < 8; ++nt) {
            const int mi = nt >> 1;
            const int q0 = (nt & 1) * 16;
            // B fragments (32x16 bf16): col = q0+lrow,
            // lanes 0-15 hold K 0..15; lanes 16-31 hold K 16..31.
            const unsigned short* bbase =
                right_t + ((size_t)(m0 + mi) * OC + (q0 + lrow)) * D_DIM + hi * 16;
            v8f c = {};
#pragma unroll
            for (int kc = 0; kc < 4; ++kc) {
                const unsigned short* p = bbase + kc * 32;
                v16bf bfrag = load_frag(p, p + 8);
                c = __builtin_amdgcn_wmma_f32_16x16x32_bf16(
                        false, afrag[kc], false, bfrag, (short)0, c, false, false);
            }
            // D tile (16x16 f32): VGPR r = row r (lanes 0-15) / r+8 (16-31),
            // col = lrow. Scatter as bf16 into LDS O[li][c][mi][q].
#pragma unroll
            for (int r = 0; r < 8; ++r) {
                int row = r + hi * 8;       // c offset within tile
                int idx = ((li * 32 + c0 + row) * 4 + mi) * 32 + q0 + lrow;
                O[idx] = f32_to_bf16(c[r]);
            }
        }
    }
    wait_wout_tile(wave);    // Wout tile 0 DMA complete (wave 0)
    __syncthreads();         // O tile + Wout tile 0 visible to all waves

    // ----------------- Stage B: output projection GEMM ---------------------
    {
        const int pbase = wave * 32;        // this wave's p-slice
        const int li = lrow >> 2;           // M row = li*4+mi
        const int mi = lrow & 3;
        v8f acc0 = {}, acc1 = {};
        for (int c = 0; c < 32; ++c) {
            // Prefetch next Wout tile into the other buffer (TDM, async).
            if (c + 1 < 32)
                load_wout_tile(WTILE[(c + 1) & 1], wout_bf, c + 1, wave);

            // A fragment from LDS: row = (li,mi), K = q.
            const unsigned short* ap =
                O + (((li * 32 + c) * 4 + mi) * 32) + hi * 16;
            v16bf af = load_frag(ap, ap + 8);
            // B fragments from LDS Wout tile: col = p, K = q (fixed c).
            const unsigned short* wt = WTILE[c & 1];
            const unsigned short* bp0 = wt + (pbase + lrow) * 32 + hi * 16;
            v16bf bf0 = load_frag(bp0, bp0 + 8);
            const unsigned short* bp1 = wt + (pbase + 16 + lrow) * 32 + hi * 16;
            v16bf bf1 = load_frag(bp1, bp1 + 8);
            acc0 = __builtin_amdgcn_wmma_f32_16x16x32_bf16(
                       false, af, false, bf0, (short)0, acc0, false, false);
            acc1 = __builtin_amdgcn_wmma_f32_16x16x32_bf16(
                       false, af, false, bf1, (short)0, acc1, false, false);

            wait_wout_tile(wave);   // tile c+1 DMA complete (wave 0)
            __syncthreads();        // all waves done with tile c; c+1 visible
        }
        // Epilogue: add bias, store f32 to out[l][m][p].
        const int p0 = pbase + lrow;
        const int p1 = pbase + 16 + lrow;
        const float b0 = bout[p0];
        const float b1 = bout[p1];
#pragma unroll
        for (int r = 0; r < 8; ++r) {
            int row = r + hi * 8;           // lm row = li*4+mi
            int l = l0 + (row >> 2);
            int m = m0 + (row & 3);
            size_t o = ((size_t)l * L_DIM + m) * PC;
            out[o + p0] = acc0[r] + b0;
            out[o + p1] = acc1[r] + b1;
        }
    }
}

// ---------------------------------------------------------------------------
extern "C" void kernel_launch(void* const* d_in, const int* in_sizes, int n_in,
                              void* d_out, int out_size, void* d_ws, size_t ws_size,
                              hipStream_t stream) {
    const float* msa  = (const float*)d_in[0];
    const float* ln_w = (const float*)d_in[1];
    const float* ln_b = (const float*)d_in[2];
    const float* Wl   = (const float*)d_in[3];
    const float* bl   = (const float*)d_in[4];
    const float* Wr   = (const float*)d_in[5];
    const float* br   = (const float*)d_in[6];
    const float* Wout = (const float*)d_in[7];
    const float* bout = (const float*)d_in[8];
    float* out = (float*)d_out;

    unsigned short* left_t  = (unsigned short*)d_ws;                  // 2 MB
    unsigned short* right_t = left_t + (size_t)L_DIM * OC * D_DIM;    // 2 MB
    unsigned short* wout_bf = right_t + (size_t)L_DIM * OC * D_DIM;   // 512 KB

    // Wout f32 -> bf16
    k_cvt_wout<<<(PC * KDIM + 255) / 256, 256, 0, stream>>>(Wout, wout_bf, PC * KDIM);

    // LayerNorm + projections: 128*256 rows, 8 rows (waves) per block
    k_ln_proj<<<(D_DIM * L_DIM) / 8, 256, 0, stream>>>(
        msa, ln_w, ln_b, Wl, bl, Wr, br, left_t, right_t);

    // Fused outer-product + projection: (256/4)^2 = 4096 blocks
    k_outer_proj<<<(L_DIM / 4) * (L_DIM / 4), 256, 0, stream>>>(
        left_t, right_t, wout_bf, bout, out);
}
